// GATClassifier_34273839022341
// MI455X (gfx1250) — compile-verified, matrix operand order checked by reading, not run on hardware
//
#include <hip/hip_runtime.h>
#include <math.h>

typedef __attribute__((ext_vector_type(16))) __bf16 v16bf;
typedef __attribute__((ext_vector_type(8)))  float  v8f;

#define TPB 256

__device__ __forceinline__ unsigned short f2bf(float f) {
    unsigned int u = __float_as_uint(f);
    unsigned int r = (u + 0x7FFFu + ((u >> 16) & 1u)) >> 16;
    return (unsigned short)r;
}

__device__ __forceinline__ void atomicMaxF32(float* addr, float v) {
    if (v >= 0.0f) atomicMax((int*)addr, __float_as_int(v));
    else atomicMin((unsigned int*)addr, (unsigned int)__float_as_int(v));
}

// ---------------- init / cast helpers ----------------

__global__ void fill_f32(float* __restrict__ p, float v, int n) {
    int i = blockIdx.x * blockDim.x + threadIdx.x;
    if (i < n) p[i] = v;
}

__global__ void cast_bf16(const float* __restrict__ src, unsigned short* __restrict__ dst, int n) {
    int i = blockIdx.x * blockDim.x + threadIdx.x;
    if (i < n) dst[i] = f2bf(src[i]);
}

// Wt[o*K + i] = W[i*ldw + o]  (o < ncols real, else 0-pad), Wt is [npad, K] bf16
__global__ void transpose_cast_bf16(const float* __restrict__ W, unsigned short* __restrict__ Wt,
                                    int K, int ldw, int ncols, int npad) {
    int i = blockIdx.x * blockDim.x + threadIdx.x;
    if (i >= npad * K) return;
    int o = i / K, k = i % K;
    Wt[(size_t)o * K + k] = (o < ncols) ? f2bf(W[(size_t)k * ldw + o]) : (unsigned short)0;
}

// ---------------- WMMA GEMM: C[M,ldc] = A[M,K] * Bt[nt*16,K]^T  (bf16 in, f32 out) ----------------
// one 16x16 output tile per wave; wave-uniform early-out keeps EXEC all-ones for v_wmma

__global__ void wmma_gemm_bf16(const unsigned short* __restrict__ A,
                               const unsigned short* __restrict__ Bt,
                               float* __restrict__ C,
                               int K, int ldc, int ntilesN, int ncols, int totalTiles) {
    int wave = threadIdx.x >> 5;
    int tile = blockIdx.x * (blockDim.x >> 5) + wave;
    if (tile >= totalTiles) return;                    // uniform per wave
    int mt = tile / ntilesN, nt = tile % ntilesN;
    int lane = threadIdx.x & 31;
    int r = lane & 15, half = lane >> 4;

    const unsigned short* aRow = A  + (size_t)(mt * 16 + r) * K;
    const unsigned short* bRow = Bt + (size_t)(nt * 16 + r) * K;

    v8f acc = {};
    for (int kk = 0; kk < K; kk += 32) {
        __builtin_prefetch(aRow + kk + 64, 0, 0);      // global_prefetch_b8
        __builtin_prefetch(bRow + kk + 64, 0, 0);
        union { unsigned int u[8]; v16bf v; } fa, fb;
#pragma unroll
        for (int i = 0; i < 4; ++i) {
            int k0 = kk + half * 8 + 2 * i;            // ISA 16-bit A layout (16x32)
            fa.u[i]     = *(const unsigned int*)(aRow + k0);
            fa.u[4 + i] = *(const unsigned int*)(aRow + k0 + 16);
            fb.u[i]     = *(const unsigned int*)(bRow + k0);
            fb.u[4 + i] = *(const unsigned int*)(bRow + k0 + 16);
        }
        acc = __builtin_amdgcn_wmma_f32_16x16x32_bf16(
                  false, fa.v, false, fb.v, (short)0, acc, false, false);
    }

    int col = nt * 16 + r;
    if (col < ncols) {
#pragma unroll
        for (int i = 0; i < 8; ++i) {                  // C layout: lane half picks M+8
            int row = mt * 16 + half * 8 + i;
            C[(size_t)row * ldc + col] = acc[i];
        }
    }
}

// ---------------- layer-1 attention ----------------

__global__ void att_scores1(const float* __restrict__ h, const float* __restrict__ asrc,
                            const float* __restrict__ adst, float* __restrict__ os,
                            float* __restrict__ od, int N) {
    int idx = blockIdx.x * blockDim.x + threadIdx.x;
    if (idx >= N * 4) return;
    int n = idx >> 2, hd = idx & 3;
    const float* hp = h + (size_t)n * 256 + hd * 64;
    const float* s = asrc + hd * 64;
    const float* d = adst + hd * 64;
    float ss = 0.f, dd = 0.f;
#pragma unroll 8
    for (int c = 0; c < 64; ++c) { float v = hp[c]; ss += v * s[c]; dd += v * d[c]; }
    os[idx] = ss; od[idx] = dd;
}

__global__ void edge1_max(const int* __restrict__ ei, int E, int Etot,
                          const float* __restrict__ as, const float* __restrict__ ad,
                          float* __restrict__ ev, float* __restrict__ nmax) {
    int e = blockIdx.x * blockDim.x + threadIdx.x;
    if (e >= Etot) return;
    int src = (e < E) ? ei[e]     : (e - E);
    int dst = (e < E) ? ei[E + e] : (e - E);
#pragma unroll
    for (int h = 0; h < 4; ++h) {
        float v = as[src * 4 + h] + ad[dst * 4 + h];
        v = (v > 0.f) ? v : 0.2f * v;                  // leaky_relu
        ev[e * 4 + h] = v;
        atomicMaxF32(&nmax[dst * 4 + h], v);
    }
}

__global__ void edge1_expsum(const int* __restrict__ ei, int E, int Etot,
                             const float* __restrict__ nmax, float* __restrict__ ev,
                             float* __restrict__ den) {
    int e = blockIdx.x * blockDim.x + threadIdx.x;
    if (e >= Etot) return;
    int dst = (e < E) ? ei[E + e] : (e - E);
#pragma unroll
    for (int h = 0; h < 4; ++h) {
        float a = expf(ev[e * 4 + h] - nmax[dst * 4 + h]);
        ev[e * 4 + h] = a;
        atomicAdd(&den[dst * 4 + h], a);
    }
}

__global__ void edge1_norm(const int* __restrict__ ei, int E, int Etot,
                           const float* __restrict__ den, float* __restrict__ ev,
                           float* __restrict__ alpha_out) {
    int idx = blockIdx.x * blockDim.x + threadIdx.x;
    if (idx >= Etot * 4) return;
    int e = idx >> 2, h = idx & 3;
    int dst = (e < E) ? ei[E + e] : (e - E);
    float a = ev[idx] / (den[dst * 4 + h] + 1e-16f);
    ev[idx] = a;
    alpha_out[idx] = a;                                // second return value of reference
}

__global__ void aggregate1(const int* __restrict__ ei, int E, int Etot,
                           const float* __restrict__ alpha, const float* __restrict__ h,
                           float* __restrict__ agg) {
    int idx = blockIdx.x * blockDim.x + threadIdx.x;
    if (idx >= Etot * 64) return;
    int e = idx >> 6, q = idx & 63;
    int c0 = q * 4, hd = c0 >> 6;
    int src = (e < E) ? ei[e]     : (e - E);
    int dst = (e < E) ? ei[E + e] : (e - E);
    float a = alpha[e * 4 + hd];
    const float4 hv = *(const float4*)(h + (size_t)src * 256 + c0);
    float* ap = agg + (size_t)dst * 256 + c0;
    atomicAdd(ap + 0, a * hv.x);
    atomicAdd(ap + 1, a * hv.y);
    atomicAdd(ap + 2, a * hv.z);
    atomicAdd(ap + 3, a * hv.w);
}

__global__ void bias_elu_cast(const float* __restrict__ agg, const float* __restrict__ b,
                              unsigned short* __restrict__ hb, int total) {
    int idx = blockIdx.x * blockDim.x + threadIdx.x;
    if (idx >= total) return;
    float v = agg[idx] + b[idx & 255];
    v = (v > 0.f) ? v : (expf(v) - 1.f);               // elu
    hb[idx] = f2bf(v);
}

// ---------------- layer-2 attention (1 head, C=40) ----------------

__global__ void att_scores2(const float* __restrict__ h2, const float* __restrict__ as,
                            const float* __restrict__ ad, float* __restrict__ os,
                            float* __restrict__ od, int N) {
    int n = blockIdx.x * blockDim.x + threadIdx.x;
    if (n >= N) return;
    const float* p = h2 + (size_t)n * 40;
    float ss = 0.f, dd = 0.f;
#pragma unroll 8
    for (int c = 0; c < 40; ++c) { float v = p[c]; ss += v * as[c]; dd += v * ad[c]; }
    os[n] = ss; od[n] = dd;
}

__global__ void edge2_max(const int* __restrict__ ei, int E, int Etot,
                          const float* __restrict__ as, const float* __restrict__ ad,
                          float* __restrict__ ev, float* __restrict__ nmax) {
    int e = blockIdx.x * blockDim.x + threadIdx.x;
    if (e >= Etot) return;
    int src = (e < E) ? ei[e]     : (e - E);
    int dst = (e < E) ? ei[E + e] : (e - E);
    float v = as[src] + ad[dst];
    v = (v > 0.f) ? v : 0.2f * v;
    ev[e] = v;
    atomicMaxF32(&nmax[dst], v);
}

__global__ void edge2_expsum(const int* __restrict__ ei, int E, int Etot,
                             const float* __restrict__ nmax, float* __restrict__ ev,
                             float* __restrict__ den) {
    int e = blockIdx.x * blockDim.x + threadIdx.x;
    if (e >= Etot) return;
    int dst = (e < E) ? ei[E + e] : (e - E);
    float a = expf(ev[e] - nmax[dst]);
    ev[e] = a;
    atomicAdd(&den[dst], a);
}

__global__ void aggregate2(const int* __restrict__ ei, int E, int Etot,
                           const float* __restrict__ ev, const float* __restrict__ den,
                           const float* __restrict__ h2, float* __restrict__ agg) {
    int idx = blockIdx.x * blockDim.x + threadIdx.x;
    if (idx >= Etot * 40) return;
    int e = idx / 40, c = idx % 40;
    int src = (e < E) ? ei[e]     : (e - E);
    int dst = (e < E) ? ei[E + e] : (e - E);
    float a = ev[e] / (den[dst] + 1e-16f);
    atomicAdd(agg + (size_t)dst * 40 + c, a * h2[(size_t)src * 40 + c]);
}

__global__ void logsoftmax_out(const float* __restrict__ agg, const float* __restrict__ b,
                               float* __restrict__ out, int N) {
    int n = blockIdx.x * blockDim.x + threadIdx.x;
    if (n >= N) return;
    const float* p = agg + (size_t)n * 40;
    float m = -__builtin_inff();
#pragma unroll 8
    for (int c = 0; c < 40; ++c) m = fmaxf(m, p[c] + b[c]);
    float s = 0.f;
#pragma unroll 8
    for (int c = 0; c < 40; ++c) s += expf(p[c] + b[c] - m);
    float l = logf(s) + m;
#pragma unroll 8
    for (int c = 0; c < 40; ++c) out[(size_t)n * 40 + c] = p[c] + b[c] - l;
}

// ---------------- launcher ----------------

static inline int cdiv(long long a, long long b) { return (int)((a + b - 1) / b); }

extern "C" void kernel_launch(void* const* d_in, const int* in_sizes, int n_in,
                              void* d_out, int out_size, void* d_ws, size_t ws_size,
                              hipStream_t stream) {
    const float* x        = (const float*)d_in[0];
    const int*   ei       = (const int*)  d_in[1];
    const float* W1       = (const float*)d_in[2];
    const float* att_src1 = (const float*)d_in[3];
    const float* att_dst1 = (const float*)d_in[4];
    const float* b1       = (const float*)d_in[5];
    const float* W2       = (const float*)d_in[6];
    const float* att_src2 = (const float*)d_in[7];
    const float* att_dst2 = (const float*)d_in[8];
    const float* b2       = (const float*)d_in[9];

    const int F    = 256;                 // input features
    const int HD1  = 256;                 // heads*H = 4*64
    const int C2   = 40;                  // layer-2 channels
    const int N    = in_sizes[0] / F;     // 50000
    const int E    = in_sizes[1] / 2;     // 800000
    const int Etot = E + N;               // with self-loops

    // workspace bump allocator (256B aligned)
    char* ws = (char*)d_ws;
    size_t off = 0;
    auto alloc = [&](size_t bytes) -> void* {
        off = (off + 255) & ~(size_t)255;
        void* p = ws + off;
        off += bytes;
        return p;
    };
    unsigned short* x_bf   = (unsigned short*)alloc((size_t)N * F * 2);
    unsigned short* W1t    = (unsigned short*)alloc((size_t)HD1 * F * 2);
    unsigned short* W2t    = (unsigned short*)alloc((size_t)48 * HD1 * 2);
    float* h1raw = (float*)alloc((size_t)N * HD1 * 4);
    float* as1   = (float*)alloc((size_t)N * 4 * 4);
    float* ad1   = (float*)alloc((size_t)N * 4 * 4);
    float* nmax1 = (float*)alloc((size_t)N * 4 * 4);
    float* den1  = (float*)alloc((size_t)N * 4 * 4);
    float* ev1   = (float*)alloc((size_t)Etot * 4 * 4);
    float* agg1  = (float*)alloc((size_t)N * HD1 * 4);
    unsigned short* h1_bf = (unsigned short*)alloc((size_t)N * HD1 * 2);
    float* h2raw = (float*)alloc((size_t)N * C2 * 4);
    float* as2   = (float*)alloc((size_t)N * 4);
    float* ad2   = (float*)alloc((size_t)N * 4);
    float* nmax2 = (float*)alloc((size_t)N * 4);
    float* den2  = (float*)alloc((size_t)N * 4);
    float* ev2   = (float*)alloc((size_t)Etot * 4);
    float* agg2  = (float*)alloc((size_t)N * C2 * 4);
    (void)ws_size;

    float* out_ls    = (float*)d_out;            // [N,40] log_softmax
    float* out_alpha = out_ls + (size_t)N * C2;  // [Etot,4] alpha (layer 1)

    const float NEG_INF = -__builtin_inff();

    // init mutable state (harness does not re-poison between replays)
    fill_f32<<<cdiv(N * 4, TPB), TPB, 0, stream>>>(nmax1, NEG_INF, N * 4);
    fill_f32<<<cdiv(N * 4, TPB), TPB, 0, stream>>>(den1, 0.f, N * 4);
    fill_f32<<<cdiv((long long)N * HD1, TPB), TPB, 0, stream>>>(agg1, 0.f, N * HD1);
    fill_f32<<<cdiv(N, TPB), TPB, 0, stream>>>(nmax2, NEG_INF, N);
    fill_f32<<<cdiv(N, TPB), TPB, 0, stream>>>(den2, 0.f, N);
    fill_f32<<<cdiv(N * C2, TPB), TPB, 0, stream>>>(agg2, 0.f, N * C2);

    // casts
    cast_bf16<<<cdiv((long long)N * F, TPB), TPB, 0, stream>>>(x, x_bf, N * F);
    transpose_cast_bf16<<<cdiv(HD1 * F, TPB), TPB, 0, stream>>>(W1, W1t, F, HD1, HD1, HD1);
    transpose_cast_bf16<<<cdiv(48 * HD1, TPB), TPB, 0, stream>>>(W2, W2t, HD1, C2, C2, 48);

    // GEMM1: h1raw = x @ W1   (M=50000 -> 3125 mtiles, 16 ntiles)
    {
        int ntN = HD1 / 16, total = (N / 16) * ntN;
        wmma_gemm_bf16<<<cdiv(total, 8), TPB, 0, stream>>>(x_bf, W1t, h1raw, F, HD1, ntN, HD1, total);
    }

    // layer-1 attention + softmax over destinations
    att_scores1<<<cdiv(N * 4, TPB), TPB, 0, stream>>>(h1raw, att_src1, att_dst1, as1, ad1, N);
    edge1_max<<<cdiv(Etot, TPB), TPB, 0, stream>>>(ei, E, Etot, as1, ad1, ev1, nmax1);
    edge1_expsum<<<cdiv(Etot, TPB), TPB, 0, stream>>>(ei, E, Etot, nmax1, ev1, den1);
    edge1_norm<<<cdiv(Etot * 4, TPB), TPB, 0, stream>>>(ei, E, Etot, den1, ev1, out_alpha);
    aggregate1<<<cdiv((long long)Etot * 64, TPB), TPB, 0, stream>>>(ei, E, Etot, ev1, h1raw, agg1);
    bias_elu_cast<<<cdiv((long long)N * HD1, TPB), TPB, 0, stream>>>(agg1, b1, h1_bf, N * HD1);

    // GEMM2: h2raw = h1 @ W2  (40 cols padded to 48 -> 3 ntiles, column-guarded store)
    {
        int ntN = 3, total = (N / 16) * ntN;
        wmma_gemm_bf16<<<cdiv(total, 8), TPB, 0, stream>>>(h1_bf, W2t, h2raw, HD1, C2, ntN, C2, total);
    }

    // layer-2 attention + softmax + aggregation
    att_scores2<<<cdiv(N, TPB), TPB, 0, stream>>>(h2raw, att_src2, att_dst2, as2, ad2, N);
    edge2_max<<<cdiv(Etot, TPB), TPB, 0, stream>>>(ei, E, Etot, as2, ad2, ev2, nmax2);
    edge2_expsum<<<cdiv(Etot, TPB), TPB, 0, stream>>>(ei, E, Etot, nmax2, ev2, den2);
    aggregate2<<<cdiv((long long)Etot * 40, TPB), TPB, 0, stream>>>(ei, E, Etot, ev2, den2, h2raw, agg2);

    // final bias + log_softmax
    logsoftmax_out<<<cdiv(N, TPB), TPB, 0, stream>>>(agg2, b2, out_ls, N);
}
